// RNNbn_30537217474609
// MI455X (gfx1250) — matrix operand order repaired and checked by reading, not compiled, not run on hardware
//
#include <hip/hip_runtime.h>
#include <hip/hip_bf16.h>

typedef unsigned short u16;
typedef __attribute__((ext_vector_type(16))) __bf16 bf16x16;
typedef __attribute__((ext_vector_type(8)))  float  f32x8;

static constexpr int B_   = 128;
static constexpr int T_   = 512;
static constexpr int NINP = 512;
static constexpr int NHID = 1024;
static constexpr int NG   = 4 * NHID;      // 4096
static constexpr int KCAT = NINP + NHID;   // 1536
static constexpr int FC1N = 341;
static constexpr int LDS_S = 40;           // bf16 stride (16B-aligned rows, conflict pad)

__device__ __forceinline__ u16 f2bf(float x) {
  unsigned u = __float_as_uint(x);
  u += 0x7FFFu + ((u >> 16) & 1u);         // round-to-nearest-even
  return (u16)(u >> 16);
}
__device__ __forceinline__ float sigm(float x) { return 1.0f / (1.0f + __expf(-x)); }

struct U4x2 { uint4 lo, hi; };
__device__ __forceinline__ bf16x16 make_frag(const u16* p0, const u16* p1) {
  U4x2 t; t.lo = *(const uint4*)p0; t.hi = *(const uint4*)p1;
  return __builtin_bit_cast(bf16x16, t);
}

// ---------------- weight prep: Wcat[g][k] = bf16([W_ih | W_hh]); bias = b_ih+b_hh
__global__ __launch_bounds__(256) void prep_weights_kernel(
    const float* __restrict__ W_ih, const float* __restrict__ W_hh,
    const float* __restrict__ b_ih, const float* __restrict__ b_hh,
    u16* __restrict__ Wcat, float* __restrict__ bias) {
  int idx = blockIdx.x * 256 + threadIdx.x;
  if (idx < NG * KCAT) {
    int g = idx / KCAT, k = idx % KCAT;
    float v = (k < NINP) ? W_ih[g * NINP + k] : W_hh[g * NHID + (k - NINP)];
    Wcat[idx] = f2bf(v);
  }
  if (idx < NG) bias[idx] = b_ih[idx] + b_hh[idx];
}

// ---------------- state init + first A = [x_0 | h_0] in bf16
__global__ __launch_bounds__(256) void prep_state_kernel(
    const float* __restrict__ hx, const float* __restrict__ cx,
    const int* __restrict__ ids, const float* __restrict__ emb,
    float* __restrict__ h, float* __restrict__ c, float* __restrict__ accum,
    u16* __restrict__ A) {
  int idx = blockIdx.x * 256 + threadIdx.x;
  if (idx >= B_ * NHID) return;
  int b = idx / NHID, j = idx % NHID;
  h[idx] = hx[idx]; c[idx] = cx[idx]; accum[idx] = 0.0f;
  A[b * KCAT + NINP + j] = f2bf(hx[idx]);
  if (j < NINP)
    A[b * KCAT + j] = f2bf(emb[(long)ids[b * T_] * NINP + j]);
}

// ---------------- per-step GEMM: gates[128][4096] = A[128][1536] x Wcat^T + bias
// block: 256 thr (8 waves), computes full M=128 x N=32; wave w -> rows 16w..16w+15
__global__ __launch_bounds__(256) void lstm_gemm_kernel(
    const u16* __restrict__ A, const u16* __restrict__ W,
    const float* __restrict__ bias, float* __restrict__ gates) {
  __shared__ __align__(16) u16 As[B_ * LDS_S];   // [m][k]
  __shared__ __align__(16) u16 Bs[32 * LDS_S];   // [n][k] (pre-transposed in global)
  const int tid  = threadIdx.x;
  const int g0   = blockIdx.x * 32;
  const int wave = tid >> 5;
  const int lane = tid & 31;
  const int r    = lane & 15;
  const int hf   = lane >> 4;
  const int m0   = wave * 16;

  f32x8 acc0 = {}; f32x8 acc1 = {};

  for (int kc = 0; kc < KCAT; kc += 32) {
    // stage A chunk: 128x32 bf16 as 512 uint4, 2 per thread
#pragma unroll
    for (int i = 0; i < 2; ++i) {
      int chunk = tid * 2 + i;
      int m = chunk >> 2, ko = (chunk & 3) * 8;
      *(uint4*)&As[m * LDS_S + ko] = *(const uint4*)&A[m * KCAT + kc + ko];
    }
    // stage B chunk (transposed layout): Bs[n][k] = W[g0+n][kc+k]
    if (tid < 128) {
      int n = tid >> 2, ko = (tid & 3) * 8;
      *(uint4*)&Bs[n * LDS_S + ko] = *(const uint4*)&W[(g0 + n) * KCAT + kc + ko];
      if (kc + 32 < KCAT)  // keep next W chunk warm (global_prefetch_b8)
        __builtin_prefetch(&W[(g0 + n) * KCAT + kc + 32 + ko], 0, 1);
    }
    __syncthreads();

    // A fragment: 16 contiguous K starting at 16*hf (matches B)
    const u16* ar = &As[(m0 + r) * LDS_S];
    bf16x16 af = make_frag(ar + 16 * hf, ar + 16 * hf + 8);
    // B fragments: 16 contiguous K starting at 16*hf, row = column index
    const u16* b0 = &Bs[r * LDS_S + 16 * hf];
    const u16* b1 = &Bs[(16 + r) * LDS_S + 16 * hf];
    bf16x16 bf0 = make_frag(b0, b0 + 8);
    bf16x16 bf1 = make_frag(b1, b1 + 8);

    acc0 = __builtin_amdgcn_wmma_f32_16x16x32_bf16(false, af, false, bf0,
                                                   (short)0, acc0, false, false);
    acc1 = __builtin_amdgcn_wmma_f32_16x16x32_bf16(false, af, false, bf1,
                                                   (short)0, acc1, false, false);
    __syncthreads();
  }

  // C/D layout: element j -> row m0 + j + 8*hf, col r
  const float bsv0 = bias[g0 + r];
  const float bsv1 = bias[g0 + 16 + r];
#pragma unroll
  for (int j = 0; j < 8; ++j) {
    int m = m0 + j + 8 * hf;
    gates[m * NG + g0 + r]       = acc0[j] + bsv0;
    gates[m * NG + g0 + 16 + r]  = acc1[j] + bsv1;
  }
}

// ---------------- LSTM cell + masked pool accum + build next-step A operand
__global__ __launch_bounds__(256) void lstm_cell_kernel(
    const float* __restrict__ gates, const int* __restrict__ seq_len,
    const int* __restrict__ ids, const float* __restrict__ emb,
    float* __restrict__ h, float* __restrict__ c, float* __restrict__ accum,
    u16* __restrict__ A, int t) {
  int idx = blockIdx.x * 256 + threadIdx.x;
  if (idx >= B_ * NHID) return;
  int b = idx / NHID, j = idx % NHID;
  const float* g4 = gates + b * NG;
  float ig = sigm(g4[j]);
  float fg = sigm(g4[NHID + j]);
  float gg = tanhf(g4[2 * NHID + j]);
  float og = sigm(g4[3 * NHID + j]);
  float cn = fg * c[idx] + ig * gg;
  float hn = og * tanhf(cn);
  c[idx] = cn; h[idx] = hn;
  A[b * KCAT + NINP + j] = f2bf(hn);                    // h half of next A
  if (t < seq_len[b]) accum[idx] += hn;                 // ragged mean-pool numerator
  if (j < NINP && (t + 1) < T_)                         // x half of next A (fused gather)
    A[b * KCAT + j] = f2bf(emb[(long)ids[b * T_ + t + 1] * NINP + j]);
}

// ---------------- head: fc1 + BN + leakyReLU  (44 MFLOP, VALU is fine)
__global__ __launch_bounds__(256) void head_fc1_kernel(
    const float* __restrict__ accum, const int* __restrict__ seq_len,
    const float* __restrict__ fc1_w, const float* __restrict__ fc1_b,
    const float* __restrict__ bn_g, const float* __restrict__ bn_b,
    const float* __restrict__ bn_m, const float* __restrict__ bn_v,
    float* __restrict__ h1) {
  int idx = blockIdx.x * 256 + threadIdx.x;
  if (idx >= B_ * FC1N) return;
  int b = idx / FC1N, n = idx % FC1N;
  float s = 0.0f;
  const float* ab = accum + b * NHID;
  const float* wr = fc1_w + n * NHID;
  for (int k = 0; k < NHID; ++k) s = fmaf(ab[k], wr[k], s);
  float v = s / (float)seq_len[b] + fc1_b[n];
  v = (v - bn_m[n]) * rsqrtf(bn_v[n] + 1e-5f) * bn_g[n] + bn_b[n];
  v = fmaxf(v, 0.01f * v);
  h1[idx] = v;
}

// ---------------- head: fc2 + log_softmax -> out[0..255]
__global__ void head_fc2_kernel(const float* __restrict__ h1,
                                const float* __restrict__ fc2_w,
                                const float* __restrict__ fc2_b,
                                float* __restrict__ out) {
  int b = threadIdx.x;
  if (b >= B_) return;
  float l0 = fc2_b[0], l1 = fc2_b[1];
  const float* hb = h1 + b * FC1N;
  for (int n = 0; n < FC1N; ++n) {
    float v = hb[n];
    l0 = fmaf(v, fc2_w[n], l0);
    l1 = fmaf(v, fc2_w[FC1N + n], l1);
  }
  float m = fmaxf(l0, l1);
  float lse = m + logf(expf(l0 - m) + expf(l1 - m));
  out[b * 2 + 0] = l0 - lse;
  out[b * 2 + 1] = l1 - lse;
}

// ---------------- copy final h, c into out (after 256 logit floats)
__global__ __launch_bounds__(256) void copy_state_kernel(
    const float* __restrict__ h, const float* __restrict__ c,
    float* __restrict__ out) {
  int i = blockIdx.x * 256 + threadIdx.x;
  if (i >= B_ * NHID) return;
  out[2 * B_ + i] = h[i];
  out[2 * B_ + B_ * NHID + i] = c[i];
}

extern "C" void kernel_launch(void* const* d_in, const int* in_sizes, int n_in,
                              void* d_out, int out_size, void* d_ws, size_t ws_size,
                              hipStream_t stream) {
  (void)in_sizes; (void)n_in; (void)out_size; (void)ws_size;
  const int*   ids   = (const int*)d_in[0];
  const int*   slen  = (const int*)d_in[1];
  const float* hx    = (const float*)d_in[2];
  const float* cx    = (const float*)d_in[3];
  const float* emb   = (const float*)d_in[4];
  const float* W_ih  = (const float*)d_in[5];
  const float* W_hh  = (const float*)d_in[6];
  const float* b_ih  = (const float*)d_in[7];
  const float* b_hh  = (const float*)d_in[8];
  const float* fc1_w = (const float*)d_in[9];
  const float* fc1_b = (const float*)d_in[10];
  const float* fc2_w = (const float*)d_in[11];
  const float* fc2_b = (const float*)d_in[12];
  const float* bn_g  = (const float*)d_in[13];
  const float* bn_b  = (const float*)d_in[14];
  const float* bn_m  = (const float*)d_in[15];
  const float* bn_v  = (const float*)d_in[16];
  float* out = (float*)d_out;

  char* ws = (char*)d_ws;
  size_t off = 0;
  auto alloc = [&](size_t bytes) -> void* {
    void* p = ws + off;
    off = (off + bytes + 255) & ~(size_t)255;
    return p;
  };
  u16*   Wcat  = (u16*)  alloc((size_t)NG * KCAT * sizeof(u16));   // 12.6 MB (L2-resident)
  float* bias  = (float*)alloc((size_t)NG * sizeof(float));
  u16*   Abuf  = (u16*)  alloc((size_t)B_ * KCAT * sizeof(u16));   // [x_t | h] bf16
  float* gates = (float*)alloc((size_t)B_ * NG * sizeof(float));
  float* hbuf  = (float*)alloc((size_t)B_ * NHID * sizeof(float));
  float* cbuf  = (float*)alloc((size_t)B_ * NHID * sizeof(float));
  float* accum = (float*)alloc((size_t)B_ * NHID * sizeof(float));
  float* h1    = (float*)alloc((size_t)B_ * FC1N * sizeof(float));

  prep_weights_kernel<<<(NG * KCAT + 255) / 256, 256, 0, stream>>>(
      W_ih, W_hh, b_ih, b_hh, Wcat, bias);
  prep_state_kernel<<<(B_ * NHID + 255) / 256, 256, 0, stream>>>(
      hx, cx, ids, emb, hbuf, cbuf, accum, Abuf);

  for (int t = 0; t < T_; ++t) {
    lstm_gemm_kernel<<<NG / 32, 256, 0, stream>>>(Abuf, Wcat, bias, gates);
    lstm_cell_kernel<<<(B_ * NHID + 255) / 256, 256, 0, stream>>>(
        gates, slen, ids, emb, hbuf, cbuf, accum, Abuf, t);
  }

  head_fc1_kernel<<<(B_ * FC1N + 255) / 256, 256, 0, stream>>>(
      accum, slen, fc1_w, fc1_b, bn_g, bn_b, bn_m, bn_v, h1);
  head_fc2_kernel<<<1, 128, 0, stream>>>(h1, fc2_w, fc2_b, out);
  copy_state_kernel<<<(B_ * NHID + 255) / 256, 256, 0, stream>>>(hbuf, cbuf, out);
}
